// RGCNModel_52561809768841
// MI455X (gfx1250) — matrix operand order repaired and checked by reading, not compiled, not run on hardware
//
#include <hip/hip_runtime.h>
#include <hip/hip_bf16.h>
#include <stdint.h>

// ---------------- problem constants (match reference) ----------------
#define HID      128
#define ENC_IN   64
#define N_ENC    20000
#define N_DIAG   50000
#define N_MED    20000
#define N_PROC   10000
#define NTOT     (N_ENC + N_DIAG + N_MED + N_PROC)   // 100000
#define OFF_DIAG N_ENC
#define OFF_MED  (N_ENC + N_DIAG)
#define OFF_PROC (N_ENC + N_DIAG + N_MED)
#define RREL     6
#define E_D      400000
#define E_M      300000
#define E_P      200000

typedef __attribute__((ext_vector_type(16))) __bf16    v16bf;
typedef __attribute__((ext_vector_type(8)))  float     v8f;
typedef __attribute__((ext_vector_type(8)))  uint32_t  v8u;
typedef __attribute__((ext_vector_type(4)))  uint32_t  u32x4;
typedef __attribute__((ext_vector_type(8)))  int32_t   i32x8;
typedef __attribute__((ext_vector_type(4)))  int32_t   i32x4;

#if __has_builtin(__builtin_amdgcn_tensor_load_to_lds)
#define HAVE_TDM 1
#else
#define HAVE_TDM 0
#endif

__device__ __forceinline__ uint16_t f2bf_bits(float f) {
  uint32_t u = __builtin_bit_cast(uint32_t, f);
  u += 0x7FFFu + ((u >> 16) & 1u);            // round-to-nearest-even
  return (uint16_t)(u >> 16);
}
__device__ __forceinline__ float bf2f(uint16_t h) {
  uint32_t u = ((uint32_t)h) << 16;
  return __builtin_bit_cast(float, u);
}

// ---------------- tiled bf16 WMMA GEMM with fused epilogue ----------------
// C[M,128] = A[M,K] x W[K,128](f32 -> bf16 on stage)
// A source: either Ab (bf16 row-major, staged via Tensor Data Mover, K=128)
//           or     Af (f32 row-major, converted while staging, any K)
// epilogue: +agg (per global row), +bias, relu; writes f32 and/or bf16.
#define LDSTR 136   // padded LDS row stride (bf16 elems): 4-bank shift per row
__global__ __launch_bounds__(256)
void gemm_bf16_wmma(const float* __restrict__ Af,
                    const uint16_t* __restrict__ Ab,
                    int M, int K, int kshift,
                    const float* __restrict__ W,
                    const float* __restrict__ bias,
                    const float* __restrict__ agg,
                    int relu,
                    float*    __restrict__ outF,
                    uint16_t* __restrict__ outB)
{
  __shared__ uint16_t As[128 * LDSTR];          // A tile: [row][k] (TDM-padded)
  __shared__ uint16_t Ws[128 * LDSTR];          // B tile transposed: [n][k]
  const int tid  = threadIdx.x;
  const int row0 = blockIdx.x * 128;

  if (Ab) {
#if HAVE_TDM
    // One TDM descriptor per block: 128x128 bf16 tile, HW-padded to stride 136.
    if (tid < 32) {
      uint64_t ga  = (uint64_t)(uintptr_t)(Ab + (size_t)row0 * HID);
      uint32_t lds = (uint32_t)(uintptr_t)(void*)As;   // low 32 bits = LDS byte addr
      int rem = M - row0; if (rem > 128) rem = 128;
      u32x4 g0;
      g0[0] = 1u;                                        // count=1, user descriptor
      g0[1] = lds;                                       // lds_addr
      g0[2] = (uint32_t)ga;                              // global_addr[31:0]
      g0[3] = (uint32_t)((ga >> 32) & 0x1FFFFFFu) | (2u << 30); // addr[56:32]|type=2
      i32x8 g1;
      g1[0] = (int)((1u << 16)      // data_size = 2 bytes
                  | (1u << 20)      // pad_enable
                  | (5u << 22)      // pad_interval: 64 DWORDs (=128 bf16 = one row)
                  | (3u << 25));    // pad_amount: 4 DWORDs (=8 bf16) -> stride 136
      g1[1] = (int)(((uint32_t)HID & 0xFFFFu) << 16);    // tensor_dim0[15:0] = 128
      g1[2] = (int)(((uint32_t)rem & 0xFFFFu) << 16);    // tensor_dim1[15:0] = rows left
      g1[3] = (int)(((uint32_t)HID & 0xFFFFu) << 16);    // tile_dim0 = 128
      g1[4] = (int)((uint32_t)128u);                     // tile_dim1 = 128
      g1[5] = (int)((uint32_t)HID);                      // tensor_dim0_stride = 128
      g1[6] = 0;
      g1[7] = 0;
      i32x4 gz4 = {0, 0, 0, 0};
      i32x8 gz8 = {0, 0, 0, 0, 0, 0, 0, 0};
      // clang-23 / amdgpu-toolchain arity: (g0, g1, g2, g3, g4, cpol)
      __builtin_amdgcn_tensor_load_to_lds(g0, g1, gz4, gz4, gz8, 0);
      __builtin_amdgcn_s_wait_tensorcnt(0);
    }
#else
    // fallback: cooperative bf16 copy with padded stride
    for (int idx = tid; idx < 128 * 64; idx += 256) {
      int r = idx >> 6, c2 = idx & 63;
      int row = row0 + r;
      uint32_t v = 0;
      if (row < M) v = ((const uint32_t*)(Ab + (size_t)row * HID))[c2];
      ((uint32_t*)(As + r * LDSTR))[c2] = v;
    }
#endif
  } else {
    // f32 A path (projection GEMM, K=64): convert while staging
    for (int idx = tid; idx < (K << 7); idx += 256) {
      int r = idx >> kshift, k = idx & (K - 1);
      int row = row0 + r;
      float v = (row < M) ? Af[(size_t)row * K + k] : 0.0f;
      As[r * LDSTR + k] = f2bf_bits(v);
    }
  }

  // stage W (K x 128 row-major) transposed so K-pairs are contiguous
  for (int idx = tid; idx < (K << 7); idx += 256) {
    int k = idx >> 7, n = idx & 127;
    Ws[n * LDSTR + k] = f2bf_bits(W[idx]);
  }
  __syncthreads();

  const int wave = tid >> 5, lane = tid & 31;
  const int half = lane >> 4, m16 = lane & 15;
  const int mrow = wave * 16 + m16;             // A row this lane feeds
  v8f acc[8] = {};

  for (int ks = 0; ks < K; ks += 32) {
    // A frag, ISA 16x32 bf16 layout: lanes0-15 K=0-7,16-23 ; lanes16-31 K=8-15,24-31
    const uint32_t* Arow = (const uint32_t*)(As + mrow * LDSTR + ks);
    v8u a;
#pragma unroll
    for (int p = 0; p < 4; ++p) a[p]     = Arow[half * 4 + p];
#pragma unroll
    for (int p = 0; p < 4; ++p) a[4 + p] = Arow[8 + half * 4 + p];
    v16bf av = __builtin_bit_cast(v16bf, a);
#pragma unroll
    for (int nt = 0; nt < 8; ++nt) {
      // B frag: N = lane&15, lanes0-15 K=0-7,16-23 ; lanes16-31 K=8-15,24-31 (pairs contiguous)
      const uint32_t* Bcol = (const uint32_t*)(Ws + (nt * 16 + m16) * LDSTR + ks);
      v8u b;
#pragma unroll
      for (int p = 0; p < 8; ++p) b[p] = Bcol[(p >> 2) * 8 + half * 4 + (p & 3)];
      v16bf bv = __builtin_bit_cast(v16bf, b);
      acc[nt] = __builtin_amdgcn_wmma_f32_16x16x32_bf16(
          false, av, false, bv, (short)0, acc[nt], false, false);
    }
  }

  // C layout: VGPR c -> M=c (lanes0-15) / M=c+8 (lanes16-31); N = lane&15
#pragma unroll
  for (int nt = 0; nt < 8; ++nt) {
    int col = nt * 16 + m16;
#pragma unroll
    for (int c = 0; c < 8; ++c) {
      int gr = row0 + wave * 16 + half * 8 + c;
      if (gr < M) {
        float v = acc[nt][c];
        if (agg)  v += agg[(size_t)gr * HID + col];
        if (bias) v += bias[col];
        if (relu) v = fmaxf(v, 0.0f);
        if (outF) outF[(size_t)gr * HID + col] = v;
        if (outB) outB[(size_t)gr * HID + col] = f2bf_bits(v);
      }
    }
  }
}

// ---------------- small utility kernels ----------------
__global__ void zero_f32(float* __restrict__ p, int n) {
  int i = blockIdx.x * 256 + threadIdx.x;
  if (i < n) p[i] = 0.0f;
}

__global__ void embed_gather_bf16(const float* __restrict__ emb, const int* __restrict__ idx,
                                  int nrows, uint16_t* __restrict__ dst) {
  int i = blockIdx.x * 256 + threadIdx.x;
  if (i < nrows * HID) {
    int r = i >> 7, c = i & 127;
    dst[i] = f2bf_bits(emb[(size_t)idx[r] * HID + c]);
  }
}

__global__ void count_edges(const int* __restrict__ dstIdx, int dstOff, int E, int rel,
                            float* __restrict__ cnt) {
  int i = blockIdx.x * 256 + threadIdx.x;
  if (i < E)
    __hip_atomic_fetch_add(&cnt[(size_t)(dstIdx[i] + dstOff) * RREL + rel], 1.0f,
                           __ATOMIC_RELAXED, __HIP_MEMORY_SCOPE_AGENT);
}

__global__ void invert_cnt(float* __restrict__ cnt, int n) {
  int i = blockIdx.x * 256 + threadIdx.x;
  if (i < n) cnt[i] = 1.0f / fmaxf(cnt[i], 1.0f);
}

// one wave32 per edge: gather bf16 row of H, scale by 1/deg(dst,rel), f32 atomic add
__global__ __launch_bounds__(256)
void scatter_edges(const uint16_t* __restrict__ Hblk,
                   const int* __restrict__ srcIdx,
                   const int* __restrict__ dstIdx, int dstOff,
                   int E, int rel,
                   const float* __restrict__ invc,
                   float* __restrict__ agg)
{
  int wave = threadIdx.x >> 5, lane = threadIdx.x & 31;
  int e = blockIdx.x * 8 + wave;
  if (e >= E) return;
  int s = srcIdx[e];
  int d = dstIdx[e] + dstOff;
  float w = invc[(size_t)d * RREL + rel];
  const uint32_t* hp = (const uint32_t*)(Hblk + (size_t)s * HID);
  uint32_t p0 = hp[lane * 2], p1 = hp[lane * 2 + 1];     // 4 bf16 per lane (b64 load)
  float f0 = bf2f((uint16_t)(p0 & 0xffff)) * w;
  float f1 = bf2f((uint16_t)(p0 >> 16))    * w;
  float f2 = bf2f((uint16_t)(p1 & 0xffff)) * w;
  float f3 = bf2f((uint16_t)(p1 >> 16))    * w;
  float* ap = agg + (size_t)d * HID + lane * 4;
  __hip_atomic_fetch_add(ap + 0, f0, __ATOMIC_RELAXED, __HIP_MEMORY_SCOPE_AGENT);
  __hip_atomic_fetch_add(ap + 1, f1, __ATOMIC_RELAXED, __HIP_MEMORY_SCOPE_AGENT);
  __hip_atomic_fetch_add(ap + 2, f2, __ATOMIC_RELAXED, __HIP_MEMORY_SCOPE_AGENT);
  __hip_atomic_fetch_add(ap + 3, f3, __ATOMIC_RELAXED, __HIP_MEMORY_SCOPE_AGENT);
}

// classifier: out[i] = dot(h[i,:128], Wc) + bc   (one wave per row)
__global__ __launch_bounds__(256)
void classify(const float* __restrict__ h, const float* __restrict__ Wc,
              const float* __restrict__ bc, float* __restrict__ out, int M)
{
  int wave = threadIdx.x >> 5, lane = threadIdx.x & 31;
  int row = blockIdx.x * 8 + wave;
  if (row >= M) return;
  const float* hr = h + (size_t)row * HID;
  float s = 0.0f;
#pragma unroll
  for (int j = 0; j < 4; ++j) s += hr[lane + 32 * j] * Wc[lane + 32 * j];
  for (int off = 16; off > 0; off >>= 1) s += __shfl_down(s, off, 32);
  if (lane == 0) out[row] = s + bc[0];
}

// ---------------- host orchestration ----------------
static inline int cdiv(int a, int b) { return (a + b - 1) / b; }

extern "C" void kernel_launch(void* const* d_in, const int* in_sizes, int n_in,
                              void* d_out, int out_size, void* d_ws, size_t ws_size,
                              hipStream_t stream) {
  const float* x_enc    = (const float*)d_in[0];
  const int*   idx_diag = (const int*)  d_in[1];
  const int*   idx_med  = (const int*)  d_in[2];
  const int*   idx_proc = (const int*)  d_in[3];
  const int*   ed_src   = (const int*)  d_in[4];
  const int*   ed_dst   = (const int*)  d_in[5];
  const int*   em_src   = (const int*)  d_in[6];
  const int*   em_dst   = (const int*)  d_in[7];
  const int*   ep_src   = (const int*)  d_in[8];
  const int*   ep_dst   = (const int*)  d_in[9];
  const float* W_proj   = (const float*)d_in[10];
  const float* b_proj   = (const float*)d_in[11];
  const float* emb_diag = (const float*)d_in[12];
  const float* emb_med  = (const float*)d_in[13];
  const float* emb_proc = (const float*)d_in[14];
  const float* W_rel1   = (const float*)d_in[15];
  const float* W_root1  = (const float*)d_in[16];
  const float* b1       = (const float*)d_in[17];
  const float* W_rel2   = (const float*)d_in[18];
  const float* W_root2  = (const float*)d_in[19];
  const float* b2       = (const float*)d_in[20];
  const float* W_cls    = (const float*)d_in[21];
  const float* b_cls    = (const float*)d_in[22];
  float* out = (float*)d_out;

  // workspace carve (all chunks 256B aligned by construction)
  char* ws = (char*)d_ws;
  uint16_t* Xb = (uint16_t*)ws; ws += sizeof(uint16_t) * (size_t)NTOT * HID; // 25.6 MB
  uint16_t* Yb = (uint16_t*)ws; ws += sizeof(uint16_t) * (size_t)NTOT * HID; // 25.6 MB
  float*    Hf = (float*)ws;    ws += sizeof(float)    * (size_t)NTOT * HID; // 51.2 MB
  float*    agg= (float*)ws;    ws += sizeof(float)    * (size_t)NTOT * HID; // 51.2 MB
  float*    inv= (float*)ws;    ws += sizeof(float)    * (size_t)NTOT * RREL;//  2.4 MB
  uint16_t* Hb = (uint16_t*)ws;                                              // 35.8 MB

  // relation -> source block (only these rows need the per-relation transform)
  const int base[RREL]   = {0, OFF_DIAG, 0, OFF_MED, 0, OFF_PROC};
  const int mrows[RREL]  = {N_ENC, N_DIAG, N_ENC, N_MED, N_ENC, N_PROC};
  const int hoff[RREL]   = {0, N_ENC, N_ENC + N_DIAG, N_ENC + N_DIAG + N_MED,
                            N_ENC + N_DIAG + N_MED + N_ENC,
                            N_ENC + N_DIAG + N_MED + 2 * N_ENC};
  const int* esrc[RREL]  = {ed_src, ed_dst, em_src, em_dst, ep_src, ep_dst};
  const int* edst[RREL]  = {ed_dst, ed_src, em_dst, em_src, ep_dst, ep_src};
  const int dstoff[RREL] = {OFF_DIAG, 0, OFF_MED, 0, OFF_PROC, 0};
  const int ecnt[RREL]   = {E_D, E_D, E_M, E_M, E_P, E_P};

  // 1) per-(dst,relation) degree -> reciprocal (folds PyG 'mean' into scatter)
  zero_f32<<<cdiv(NTOT * RREL, 256), 256, 0, stream>>>(inv, NTOT * RREL);
  for (int r = 0; r < RREL; ++r)
    count_edges<<<cdiv(ecnt[r], 256), 256, 0, stream>>>(edst[r], dstoff[r], ecnt[r], r, inv);
  invert_cnt<<<cdiv(NTOT * RREL, 256), 256, 0, stream>>>(inv, NTOT * RREL);

  // 2) build bf16 node features: encounter projection (WMMA, K=64) + embedding gathers
  gemm_bf16_wmma<<<cdiv(N_ENC, 128), 256, 0, stream>>>(
      x_enc, nullptr, N_ENC, ENC_IN, 6, W_proj, b_proj, nullptr, 0, nullptr, Xb);
  embed_gather_bf16<<<cdiv(N_DIAG * HID, 256), 256, 0, stream>>>(
      emb_diag, idx_diag, N_DIAG, Xb + (size_t)OFF_DIAG * HID);
  embed_gather_bf16<<<cdiv(N_MED * HID, 256), 256, 0, stream>>>(
      emb_med, idx_med, N_MED, Xb + (size_t)OFF_MED * HID);
  embed_gather_bf16<<<cdiv(N_PROC * HID, 256), 256, 0, stream>>>(
      emb_proc, idx_proc, N_PROC, Xb + (size_t)OFF_PROC * HID);

  // 3) RGCN layers (features in bf16; TDM stages A tiles into LDS)
  auto run_layer = [&](const uint16_t* Xin, const float* Wrel, const float* Wroot,
                       const float* b, float* XoutF, uint16_t* XoutB) {
    zero_f32<<<cdiv(NTOT * HID, 256), 256, 0, stream>>>(agg, NTOT * HID);
    for (int r = 0; r < RREL; ++r)   // H[r] = Xin[block r] @ W_rel[r]  -> bf16
      gemm_bf16_wmma<<<cdiv(mrows[r], 128), 256, 0, stream>>>(
          nullptr, Xin + (size_t)base[r] * HID, mrows[r], HID, 7,
          Wrel + (size_t)r * HID * HID, nullptr, nullptr, 0,
          nullptr, Hb + (size_t)hoff[r] * HID);
    for (int r = 0; r < RREL; ++r)   // agg[dst] += H[r][src] / deg(dst,r)
      scatter_edges<<<cdiv(ecnt[r], 8), 256, 0, stream>>>(
          Hb + (size_t)hoff[r] * HID, esrc[r], edst[r], dstoff[r], ecnt[r], r, inv, agg);
    // Xout = relu(agg + Xin @ W_root + b)
    gemm_bf16_wmma<<<cdiv(NTOT, 128), 256, 0, stream>>>(
        nullptr, Xin, NTOT, HID, 7, Wroot, b, agg, 1, XoutF, XoutB);
  };
  run_layer(Xb, W_rel1, W_root1, b1, nullptr, Yb);   // layer 1 -> bf16 features
  run_layer(Yb, W_rel2, W_root2, b2, Hf, nullptr);   // layer 2 -> f32 for classifier

  // 4) classifier over encounter rows
  classify<<<cdiv(N_ENC, 8), 256, 0, stream>>>(Hf, W_cls, b_cls, out, N_ENC);

  (void)in_sizes; (void)n_in; (void)out_size; (void)ws_size;
}
